// InteractionBlock_49838800503616
// MI455X (gfx1250) — matrix operand order repaired and checked by reading, not compiled, not run on hardware
//
#include <hip/hip_runtime.h>
#include <hip/hip_bf16.h>

// ---------------------------------------------------------------------------
// SchNet InteractionBlock for MI455X (gfx1250, wave32, WMMA)
//   h1  = h @ lin1_w^T
//   W   = ssp(edge_attr @ w0^T + b0) @ w2^T + b2        (fused edge kernel)
//   agg = segment_sum(c * W * h1[src], dst)              (fused, atomicAdd)
//   h2  = (h1 + agg) @ lin2_w^T + lin2_b
//   out = ssp(h2) @ lin_w^T + lin_b
// All GEMMs: f16 inputs / f32 accumulation via v_wmma_f32_16x16x32_f16.
// Node GEMM A tiles are staged once per block in LDS (8 waves share them).
// LDS tiles use a 136-f16 row stride so ds_loads are bank-conflict free.
// ---------------------------------------------------------------------------

#define NN   50000
#define EE   600000
#define HID  128
#define NF   128
#define NG   50
#define NGP  64          // NG padded to WMMA K granularity
#define LDS_STRIDE 136   // 128 + 8 f16 pad: 68 dwords -> conflict-free banks
#define CUTOFF 10.0f
#define PI_F 3.14159265358979323846f
#define LOG2_F 0.6931471805599453f

typedef __attribute__((ext_vector_type(16))) _Float16 v16h;
typedef __attribute__((ext_vector_type(8)))  _Float16 v8h;
typedef __attribute__((ext_vector_type(8)))  float    v8f;

__device__ __forceinline__ float sspf(float x) {
    // softplus(x) - log(2), numerically stable
    float ax = fabsf(x);
    float sp = fmaxf(x, 0.0f) + __logf(1.0f + __expf(-ax));
    return sp - LOG2_F;
}

__device__ __forceinline__ v8f wmma_f16(v16h a, v16h b, v8f c) {
    return __builtin_amdgcn_wmma_f32_16x16x32_f16(
        /*neg_a=*/false, a, /*neg_b=*/false, b,
        /*c_mod=*/(short)0, c, /*reuse_a=*/false, /*reuse_b=*/false);
}

// Build a 16x32 A fragment (contiguous k = 16*half + s order, same as B) from two 8-f16 LDS loads.
__device__ __forceinline__ v16h a_frag_from_lds(const _Float16* tl,
                                                int l16, int half, int K0) {
    v8h lo = *(const v8h*)(tl + l16 * LDS_STRIDE + K0 + half * 16);      // g=0
    v8h hi = *(const v8h*)(tl + l16 * LDS_STRIDE + K0 + half * 16 + 8);  // g=1
    v16h a;
#pragma unroll
    for (int i = 0; i < 8; ++i) { a[i] = lo[i]; a[8 + i] = hi[i]; }
    return a;
}

// ---------------------------------------------------------------------------
// Prep kernels: f32 -> f16 weight conversion ([out][k] layout == torch
// [out,in] layout), zero the aggregation buffer.
// ---------------------------------------------------------------------------
__global__ void cvt_f16_kernel(const float* __restrict__ s,
                               _Float16* __restrict__ d, int n) {
    int i = blockIdx.x * blockDim.x + threadIdx.x;
    if (i < n) d[i] = (_Float16)s[i];
}

__global__ void cvt_pad_w0_kernel(const float* __restrict__ s,
                                  _Float16* __restrict__ d) {
    int i = blockIdx.x * blockDim.x + threadIdx.x;   // over 128*64
    if (i < NF * NGP) {
        int o = i >> 6, k = i & 63;
        d[i] = (_Float16)((k < NG) ? s[o * NG + k] : 0.0f);
    }
}

__global__ void zero_f32_kernel(float* __restrict__ p, int n) {
    int i = blockIdx.x * blockDim.x + threadIdx.x;
    if (i < n) p[i] = 0.0f;
}

// ---------------------------------------------------------------------------
// Node GEMM: D = epilogue( f(A0 (+A1)) @ Wf^T ) (+bias)
//   - 256 threads = 8 waves; block owns one 16-row M tile (NN = 3125*16).
//   - A tile is cooperatively loaded/transformed/f16-converted ONCE into LDS
//     (each thread handles one 8-float segment), then all 8 waves build
//     their A fragments from LDS; each wave owns one 16-col N tile.
// ---------------------------------------------------------------------------
template <bool ADD, bool SSP, bool BIAS>
__global__ __launch_bounds__(256) void node_gemm_kernel(
    const float* __restrict__ A0, const float* __restrict__ A1,
    const _Float16* __restrict__ Wf, const float* __restrict__ bias,
    float* __restrict__ D)
{
    __shared__ __align__(16) _Float16 atile[16 * LDS_STRIDE];  // ~4.3 KB

    const int tid  = threadIdx.x;
    const int wave = tid >> 5;             // N-tile 0..7
    const int lane = tid & 31;
    const int half = lane >> 4;            // 0: lanes 0-15, 1: lanes 16-31
    const int l16  = lane & 15;
    const int mtile = blockIdx.x;

    // ---- cooperative A-tile stage: 16 rows x 128 cols, 8 f32 per thread --
    {
        const int r   = tid >> 4;          // row 0..15
        const int seg = tid & 15;          // 8-col segment 0..15
        const size_t base = (size_t)(mtile * 16 + r) * HID + seg * 8;
        v8h hv;
#pragma unroll
        for (int i = 0; i < 8; ++i) {
            float x = A0[base + i];
            if (ADD) x += A1[base + i];
            if (SSP) x = sspf(x);
            hv[i] = (_Float16)x;
        }
        *(v8h*)(atile + r * LDS_STRIDE + seg * 8) = hv;
    }
    __syncthreads();   // grid is exact: every wave of every block reaches this

    // ---- per-wave 16x16 tile: 4 x WMMA over K=128 ------------------------
    const _Float16* bbase = Wf + (size_t)(wave * 16 + l16) * HID;
    v8f acc = {};
#pragma unroll
    for (int kk = 0; kk < 4; ++kk) {
        const int K0 = kk * 32;
        v16h a = a_frag_from_lds(atile, l16, half, K0);
        v16h b = *(const v16h*)(bbase + K0 + half * 16);  // contiguous 32B
        acc = wmma_f16(a, b, acc);
    }

    const float bv = BIAS ? bias[wave * 16 + l16] : 0.0f;
#pragma unroll
    for (int r = 0; r < 8; ++r) {
        const int m = half * 8 + r;        // C/D layout: VGPR r -> M = r (+8)
        D[(size_t)(mtile * 16 + m) * NF + wave * 16 + l16] = acc[r] + bv;
    }
}

// ---------------------------------------------------------------------------
// Fused edge kernel. Each wave owns a 16-edge tile (EE = 37500 * 16 exactly):
//   t   = ssp(edge_attr[16x64pad] @ W0^T + b0)          (16 WMMA)
//   W   = t @ W2^T + b2                                 (32 WMMA)
//   agg[dst] += c * W * h1[src]                         (f32 atomics, L2-hot)
// GEMM1 C-layout -> GEMM2 A-layout transpose goes through a per-wave LDS
// tile (same-wave DS ops are in-order; no block barrier required).
// ---------------------------------------------------------------------------
__global__ __launch_bounds__(256) void edge_kernel(
    const float* __restrict__ edge_attr,    // [EE, NG]
    const float* __restrict__ edge_weight,  // [EE]
    const int*   __restrict__ edge_index,   // [2, EE]
    const _Float16* __restrict__ W0f,       // [NF][NGP] f16
    const float* __restrict__ b0,           // [NF]
    const _Float16* __restrict__ W2f,       // [NF][NF] f16
    const float* __restrict__ b2,           // [NF]
    const float* __restrict__ h1,           // [NN, NF]
    float* __restrict__ agg)                // [NN, NF]
{
    __shared__ __align__(16) _Float16 lds_t[8][16 * LDS_STRIDE]; // ~34.8 KB
    __shared__ float lds_c[8][16];
    __shared__ int   lds_src[8][16];
    __shared__ int   lds_dst[8][16];

    const int wave = threadIdx.x >> 5;
    const int lane = threadIdx.x & 31;
    const int half = lane >> 4;
    const int l16  = lane & 15;

    int tile = blockIdx.x * 8 + wave;
    const bool active = (tile < (EE / 16));
    if (!active) tile = 0;                  // keep EXEC all-ones for WMMA
    const int e0 = tile * 16;

    // Stream hint for the next block's edge_attr slab (global_prefetch_b8).
    if (e0 + 144 <= EE)
        __builtin_prefetch(edge_attr + (size_t)(e0 + 128) * NG + lane * 16, 0, 1);

    // Per-edge scalars (cos envelope, src/dst) staged in LDS by lanes 0..15.
    if (lane < 16) {
        const int e = e0 + l16;
        const float w = edge_weight[e];
        lds_c[wave][l16]   = 0.5f * (__cosf(w * (PI_F / CUTOFF)) + 1.0f);
        lds_src[wave][l16] = edge_index[e];
        lds_dst[wave][l16] = edge_index[EE + e];
    }

    // ---- GEMM1: t = edge_attr(16 x 64pad) @ W0^T  ------------------------
    v16h a1[2];
#pragma unroll
    for (int kc = 0; kc < 2; ++kc) {
#pragma unroll
        for (int g = 0; g < 2; ++g) {
#pragma unroll
            for (int i = 0; i < 8; ++i) {
                const int k = kc * 32 + half * 16 + g * 8 + i;
                float v = (k < NG) ? edge_attr[(size_t)(e0 + l16) * NG + k]
                                   : 0.0f;
                a1[kc][g * 8 + i] = (_Float16)v;
            }
        }
    }
    v8f t[8] = {};
#pragma unroll
    for (int nt = 0; nt < 8; ++nt) {
#pragma unroll
        for (int kc = 0; kc < 2; ++kc) {
            v16h b = *(const v16h*)(W0f + (size_t)(nt * 16 + l16) * NGP
                                        + kc * 32 + half * 16);
            t[nt] = wmma_f16(a1[kc], b, t[nt]);
        }
    }

    // ---- bias + ssp, f16, C-layout -> LDS row-major [16][128] ------------
    _Float16* tl = lds_t[wave];
#pragma unroll
    for (int nt = 0; nt < 8; ++nt) {
        const float bb = b0[nt * 16 + l16];
#pragma unroll
        for (int r = 0; r < 8; ++r) {
            const int m = half * 8 + r;
            tl[m * LDS_STRIDE + nt * 16 + l16] = (_Float16)sspf(t[nt][r] + bb);
        }
    }

    // ---- A2 fragments from LDS (A layout), GEMM2: W = t @ W2^T -----------
    v16h a2[4];
#pragma unroll
    for (int kc = 0; kc < 4; ++kc)
        a2[kc] = a_frag_from_lds(tl, l16, half, kc * 32);

    v8f w[8] = {};
#pragma unroll
    for (int nt = 0; nt < 8; ++nt) {
#pragma unroll
        for (int kc = 0; kc < 4; ++kc) {
            v16h b = *(const v16h*)(W2f + (size_t)(nt * 16 + l16) * NF
                                        + kc * 32 + half * 16);
            w[nt] = wmma_f16(a2[kc], b, w[nt]);
        }
    }

    // ---- epilogue: agg[dst] += c * (W + b2) * h1[src]  -------------------
    if (active) {
#pragma unroll
        for (int nt = 0; nt < 8; ++nt) {
            const int n = nt * 16 + l16;
            const float bb = b2[n];
#pragma unroll
            for (int r = 0; r < 8; ++r) {
                const int m = half * 8 + r;
                const float Wv = w[nt][r] + bb;
                const float cc = lds_c[wave][m];
                const int s = lds_src[wave][m];
                const int d = lds_dst[wave][m];
                const float msg = cc * Wv * h1[(size_t)s * NF + n];
                atomicAdd(agg + (size_t)d * NF + n, msg);
            }
        }
    }
}

// ---------------------------------------------------------------------------
// Host launcher
// ---------------------------------------------------------------------------
extern "C" void kernel_launch(void* const* d_in, const int* in_sizes, int n_in,
                              void* d_out, int out_size, void* d_ws, size_t ws_size,
                              hipStream_t stream)
{
    (void)in_sizes; (void)n_in; (void)out_size; (void)ws_size;

    const float* h           = (const float*)d_in[0];
    const int*   edge_index  = (const int*)d_in[1];
    const float* edge_weight = (const float*)d_in[2];
    const float* edge_attr   = (const float*)d_in[3];
    const float* mlp_w0      = (const float*)d_in[4];
    const float* mlp_b0      = (const float*)d_in[5];
    const float* mlp_w2      = (const float*)d_in[6];
    const float* mlp_b2      = (const float*)d_in[7];
    const float* lin1_w      = (const float*)d_in[8];
    const float* lin2_w      = (const float*)d_in[9];
    const float* lin2_b      = (const float*)d_in[10];
    const float* lin_w       = (const float*)d_in[11];
    const float* lin_b       = (const float*)d_in[12];
    float* out = (float*)d_out;

    // Workspace carve-up (~77 MB + small f16 weights)
    char* ws = (char*)d_ws;
    size_t off = 0;
    auto carve = [&](size_t bytes) -> void* {
        void* p = ws + off;
        off = (off + bytes + 255) & ~(size_t)255;
        return p;
    };
    float*     h1     = (float*)    carve((size_t)NN * NF * 4);
    float*     agg    = (float*)    carve((size_t)NN * NF * 4);
    float*     h2     = (float*)    carve((size_t)NN * HID * 4);
    _Float16*  lin1f  = (_Float16*) carve((size_t)NF * HID * 2);
    _Float16*  lin2f  = (_Float16*) carve((size_t)HID * NF * 2);
    _Float16*  linf   = (_Float16*) carve((size_t)HID * HID * 2);
    _Float16*  w2f    = (_Float16*) carve((size_t)NF * NF * 2);
    _Float16*  w0pf   = (_Float16*) carve((size_t)NF * NGP * 2);

    // --- prep: weight conversion + zero agg ---
    cvt_f16_kernel<<<(NF * HID + 255) / 256, 256, 0, stream>>>(lin1_w, lin1f, NF * HID);
    cvt_f16_kernel<<<(HID * NF + 255) / 256, 256, 0, stream>>>(lin2_w, lin2f, HID * NF);
    cvt_f16_kernel<<<(HID * HID + 255) / 256, 256, 0, stream>>>(lin_w, linf, HID * HID);
    cvt_f16_kernel<<<(NF * NF + 255) / 256, 256, 0, stream>>>(mlp_w2, w2f, NF * NF);
    cvt_pad_w0_kernel<<<(NF * NGP + 255) / 256, 256, 0, stream>>>(mlp_w0, w0pf);
    zero_f32_kernel<<<((size_t)NN * NF + 255) / 256, 256, 0, stream>>>(agg, NN * NF);

    const int mtiles = NN / 16;           // 3125, exact
    const int etiles = EE / 16;           // 37500, exact
    const int eblocks = (etiles + 7) / 8; // 8 waves (tiles) per block

    // h1 = h @ lin1_w^T
    node_gemm_kernel<false, false, false><<<mtiles, 256, 0, stream>>>(
        h, nullptr, lin1f, nullptr, h1);

    // fused edge MLP + message + segment_sum
    edge_kernel<<<eblocks, 256, 0, stream>>>(
        edge_attr, edge_weight, edge_index, w0pf, mlp_b0, w2f, mlp_b2, h1, agg);

    // h2 = (h1 + agg) @ lin2_w^T + lin2_b
    node_gemm_kernel<true, false, true><<<mtiles, 256, 0, stream>>>(
        h1, agg, lin2f, lin2_b, h2);

    // out = ssp(h2) @ lin_w^T + lin_b
    node_gemm_kernel<false, true, true><<<mtiles, 256, 0, stream>>>(
        h2, nullptr, linf, lin_b, out);
}
